// SelfAttention_90950227460365
// MI455X (gfx1250) — matrix operand order repaired and checked by reading, not compiled, not run on hardware
//
#include <hip/hip_runtime.h>

typedef __attribute__((ext_vector_type(16))) __bf16 v16bf;
typedef __attribute__((ext_vector_type(8)))  float  v8f;

#define BB    16
#define CC    256
#define NN    1024
#define HH    8
#define DD    64
#define INNER 512
#define NQKV  1536
#define ATTN_SCALE 0.125f

// ---------- helpers ----------

// float -> bf16 (round-to-nearest-even) via bit ops
__device__ __forceinline__ __bf16 f2bf(float f) {
    union { float f; unsigned u; } in; in.f = f;
    unsigned r = (in.u + 0x7fffu + ((in.u >> 16) & 1u)) >> 16;
    union { unsigned short s; __bf16 h; } out; out.s = (unsigned short)r;
    return out.h;
}

// A-matrix fragment (16x32 bf16, M rows x K): lane r = row, elements:
//   K = hi*8 + [0..7]  and  K = hi*8 + 16 + [0..7]   (two 16B loads)
__device__ __forceinline__ v16bf load_a(const __bf16* row, int hi) {
    const __bf16* p = row + hi * 8;
    union { v16bf v; uint4 q[2]; } f;
    f.q[0] = *(const uint4*)(p);
    f.q[1] = *(const uint4*)(p + 16);
    return f.v;
}

// B-matrix fragment loaded from B^T (N rows x K): lane r = col N, elements:
//   K = hi*8 + [0..7]  and  K = hi*8 + 16 + [0..7]   (two 16B loads)
__device__ __forceinline__ v16bf load_b(const __bf16* row, int hi) {
    const __bf16* p = row + hi * 8;
    union { v16bf v; uint4 q[2]; } f;
    f.q[0] = *(const uint4*)(p);
    f.q[1] = *(const uint4*)(p + 16);
    return f.v;
}

__device__ __forceinline__ v8f wmma_bf16(v16bf a, v16bf b, v8f c) {
    return __builtin_amdgcn_wmma_f32_16x16x32_bf16(
        false, a, false, b, (short)0, c, false, false);
}

__device__ __forceinline__ v8f zero8() {
    v8f z;
    #pragma unroll
    for (int i = 0; i < 8; ++i) z[i] = 0.0f;
    return z;
}

// ---------- prep: transpose [R][C] f32 -> [C][R] bf16 ----------
__global__ __launch_bounds__(256) void transpose_to_bf16(
    const float* __restrict__ src, __bf16* __restrict__ dst,
    int R, int C, long sBatch, long dBatch)
{
    __shared__ float tile[32][33];
    src += (long)blockIdx.z * sBatch;
    dst += (long)blockIdx.z * dBatch;
    int c0 = blockIdx.x * 32, r0 = blockIdx.y * 32;
    int tx = threadIdx.x, ty = threadIdx.y;   // 32 x 8
    #pragma unroll
    for (int i = 0; i < 32; i += 8)
        tile[ty + i][tx] = src[(long)(r0 + ty + i) * C + c0 + tx];
    __syncthreads();
    #pragma unroll
    for (int i = 0; i < 32; i += 8)
        dst[(long)(c0 + ty + i) * R + r0 + tx] = f2bf(tile[tx][ty + i]);
}

// ---------- GEMM 1: qkv = xs @ Wqkv + bqkv ----------
// 32x64 tile per wave: 2 A-frags x 4 B-frags -> 8 WMMAs per k-chunk.
// A: xs_bf [16384][256], B^T: WqkvT [1536][256]. Scatter into Q,K [bh][n][64], Vt [bh][64][n].
__global__ __launch_bounds__(256) void gemm_qkv(
    const __bf16* __restrict__ A, const __bf16* __restrict__ Bt,
    const float* __restrict__ bias,
    __bf16* __restrict__ Qo, __bf16* __restrict__ Ko, __bf16* __restrict__ Vt)
{
    const int lane = threadIdx.x & 31;
    const int wid  = threadIdx.x >> 5;
    const int r = lane & 15, hi = lane >> 4;
    const int m0 = blockIdx.x * 256 + wid * 32;   // token rows (32 per wave)
    const int n0 = blockIdx.y * 64;               // cols (64 per wave)

    v8f acc[2][4];
    #pragma unroll
    for (int mi = 0; mi < 2; ++mi)
        #pragma unroll
        for (int ni = 0; ni < 4; ++ni) acc[mi][ni] = zero8();

    const __bf16* ap0 = A  + (long)(m0 + r) * CC;
    const __bf16* ap1 = A  + (long)(m0 + 16 + r) * CC;
    const __bf16* bp  = Bt + (long)(n0 + r) * CC;
    #pragma unroll
    for (int kc = 0; kc < 8; ++kc) {
        const v16bf a0 = load_a(ap0 + kc * 32, hi);
        const v16bf a1 = load_a(ap1 + kc * 32, hi);
        #pragma unroll
        for (int ni = 0; ni < 4; ++ni) {
            const v16bf bf = load_b(bp + (long)ni * 16 * CC + kc * 32, hi);
            acc[0][ni] = wmma_bf16(a0, bf, acc[0][ni]);
            acc[1][ni] = wmma_bf16(a1, bf, acc[1][ni]);
        }
    }

    // n-tile of 64 lies entirely in one Q/K/V segment of one head (192 = 3*64)
    const int h   = n0 / 192;
    const int seg = (n0 % 192) >> 6;   // 0=Q 1=K 2=V
    const int b   = m0 >> 10;
    const int t0  = m0 & 1023;
    const long bh = (long)b * HH + h;

    #pragma unroll
    for (int ni = 0; ni < 4; ++ni) {
        const int d = ni * 16 + r;
        const float bv = bias[n0 + ni * 16 + r];
        #pragma unroll
        for (int mi = 0; mi < 2; ++mi) {
            const int t = t0 + mi * 16;
            if (seg < 2) {
                __bf16* dst = (seg == 0 ? Qo : Ko) + (bh * NN + t) * DD + d;
                #pragma unroll
                for (int i = 0; i < 8; ++i)
                    dst[(long)(i + hi * 8) * DD] = f2bf(acc[mi][ni][i] + bv);
            } else {
                // V transposed: rows contiguous in t -> one b128 store of 8 bf16
                union { uint4 q; __bf16 h8[8]; } pk;
                #pragma unroll
                for (int i = 0; i < 8; ++i) pk.h8[i] = f2bf(acc[mi][ni][i] + bv);
                *(uint4*)(Vt + (bh * DD + d) * NN + t + hi * 8) = pk.q;
            }
        }
    }
}

// ---------- Flash attention ----------
// Block = 4 waves / 64 query rows. K (32x64) and V^T (64x32) tiles are staged
// in LDS cooperatively (double-buffered) and shared by all 4 waves; Q stays in
// registers for the whole j-loop.
__global__ __launch_bounds__(128) void attn_kernel(
    const __bf16* __restrict__ Q, const __bf16* __restrict__ K,
    const __bf16* __restrict__ Vt, __bf16* __restrict__ R)
{
    __shared__ __align__(16) __bf16 kbuf[2][32 * 64];
    __shared__ __align__(16) __bf16 vbuf[2][64 * 32];
    __shared__ __align__(16) __bf16 pbuf[4][16 * 32];

    const int tid  = threadIdx.x;
    const int lane = tid & 31;
    const int wid  = tid >> 5;
    const int r = lane & 15, hi = lane >> 4;
    const int bh = blockIdx.x;                     // 0..127
    const int i0 = blockIdx.y * 64 + wid * 16;     // query rows in [0,1024)

    const __bf16* qrow = Q + ((long)bh * NN + i0 + r) * DD;
    const v16bf qf0 = load_a(qrow,      hi);   // d = 0..31
    const v16bf qf1 = load_a(qrow + 32, hi);   // d = 32..63

    v8f o0 = zero8(), o1 = zero8(), o2 = zero8(), o3 = zero8();
    float mrow[8], lrow[8];
    #pragma unroll
    for (int i = 0; i < 8; ++i) { mrow[i] = -1e30f; lrow[i] = 0.0f; }

    __bf16* pb = pbuf[wid];

    // cooperative K/V tile load: 512 x uint4 over 128 threads
    const uint4* gkBase = (const uint4*)(K + (long)bh * NN * DD);
    const int vrow1 = tid >> 2, vc4 = tid & 3;     // rows for V^T staging

    #define LOAD_KV(J0, BUF)                                                      \
        do {                                                                      \
            uint4* sk = (uint4*)kbuf[BUF];                                        \
            const uint4* gk = gkBase + (long)(J0) * (DD / 8);                     \
            sk[tid]       = gk[tid];                                              \
            sk[tid + 128] = gk[tid + 128];                                        \
            uint4* sv = (uint4*)vbuf[BUF];                                        \
            sv[tid]       = *(const uint4*)(Vt + ((long)bh * DD + vrow1) * NN +   \
                                            (J0) + vc4 * 8);                      \
            sv[tid + 128] = *(const uint4*)(Vt + ((long)bh * DD + vrow1 + 32) * NN + \
                                            (J0) + vc4 * 8);                      \
        } while (0)

    LOAD_KV(0, 0);
    __syncthreads();

    for (int jb = 0; jb < 32; ++jb) {
        const int cur = jb & 1;
        if (jb + 1 < 32) LOAD_KV((jb + 1) * 32, cur ^ 1);

        const __bf16* kb = kbuf[cur];
        const __bf16* vb = vbuf[cur];

        // ---- S = Q K^T (two 16-wide N tiles) ----
        v8f s0 = zero8(), s1 = zero8();
        s0 = wmma_bf16(qf0, load_b(kb + r * 64,             hi), s0);
        s0 = wmma_bf16(qf1, load_b(kb + r * 64 + 32,        hi), s0);
        s1 = wmma_bf16(qf0, load_b(kb + (16 + r) * 64,      hi), s1);
        s1 = wmma_bf16(qf1, load_b(kb + (16 + r) * 64 + 32, hi), s1);

        // ---- online softmax (rows reduce across 16-lane half) ----
        float corr[8];
        #pragma unroll
        for (int i = 0; i < 8; ++i) {
            s0[i] *= ATTN_SCALE;
            s1[i] *= ATTN_SCALE;
            float mm = fmaxf(s0[i], s1[i]);
            #pragma unroll
            for (int m = 1; m < 16; m <<= 1) mm = fmaxf(mm, __shfl_xor(mm, m, 32));
            float mn = fmaxf(mrow[i], mm);
            corr[i] = __expf(mrow[i] - mn);
            mrow[i] = mn;
            s0[i] = __expf(s0[i] - mn);
            s1[i] = __expf(s1[i] - mn);
            float rs = s0[i] + s1[i];
            #pragma unroll
            for (int m = 1; m < 16; m <<= 1) rs += __shfl_xor(rs, m, 32);
            lrow[i] = lrow[i] * corr[i] + rs;
            o0[i] *= corr[i]; o1[i] *= corr[i]; o2[i] *= corr[i]; o3[i] *= corr[i];
        }

        // ---- P: C-layout -> per-wave LDS [16][32] -> A-fragment ----
        #pragma unroll
        for (int i = 0; i < 8; ++i) {
            pb[(i + hi * 8) * 32 + r]      = f2bf(s0[i]);
            pb[(i + hi * 8) * 32 + 16 + r] = f2bf(s1[i]);
        }
        const v16bf pf = load_a(pb + r * 32, hi);

        // ---- O += P V ----
        o0 = wmma_bf16(pf, load_b(vb + r * 32,        hi), o0);
        o1 = wmma_bf16(pf, load_b(vb + (16 + r) * 32, hi), o1);
        o2 = wmma_bf16(pf, load_b(vb + (32 + r) * 32, hi), o2);
        o3 = wmma_bf16(pf, load_b(vb + (48 + r) * 32, hi), o3);

        __syncthreads();
    }
    #undef LOAD_KV

    // ---- epilogue: normalize, store res [b][n][h*64+d] bf16 ----
    const int b = bh >> 3, h = bh & 7;
    __bf16* dst = R + ((long)b * NN + i0) * INNER + h * DD;
    #pragma unroll
    for (int i = 0; i < 8; ++i) {
        const float inv = 1.0f / lrow[i];
        const long rowoff = (long)(i + hi * 8) * INNER;
        dst[rowoff +      r] = f2bf(o0[i] * inv);
        dst[rowoff + 16 + r] = f2bf(o1[i] * inv);
        dst[rowoff + 32 + r] = f2bf(o2[i] * inv);
        dst[rowoff + 48 + r] = f2bf(o3[i] * inv);
    }
}

// ---------- GEMM 3: out = res @ Wout + bout + xs, stored as [b][c][t] ----------
// 32x64 tile per wave.
__global__ __launch_bounds__(256) void gemm_out(
    const __bf16* __restrict__ A, const __bf16* __restrict__ Bt,
    const float* __restrict__ bias, const float* __restrict__ x,
    float* __restrict__ out)
{
    const int lane = threadIdx.x & 31;
    const int wid  = threadIdx.x >> 5;
    const int r = lane & 15, hi = lane >> 4;
    const int m0 = blockIdx.x * 256 + wid * 32;   // token rows
    const int c0 = blockIdx.y * 64;               // channel cols

    v8f acc[2][4];
    #pragma unroll
    for (int mi = 0; mi < 2; ++mi)
        #pragma unroll
        for (int ni = 0; ni < 4; ++ni) acc[mi][ni] = zero8();

    const __bf16* ap0 = A  + (long)(m0 + r) * INNER;
    const __bf16* ap1 = A  + (long)(m0 + 16 + r) * INNER;
    const __bf16* bp  = Bt + (long)(c0 + r) * INNER;
    #pragma unroll
    for (int kc = 0; kc < 16; ++kc) {
        const v16bf a0 = load_a(ap0 + kc * 32, hi);
        const v16bf a1 = load_a(ap1 + kc * 32, hi);
        #pragma unroll
        for (int ni = 0; ni < 4; ++ni) {
            const v16bf bf = load_b(bp + (long)ni * 16 * INNER + kc * 32, hi);
            acc[0][ni] = wmma_bf16(a0, bf, acc[0][ni]);
            acc[1][ni] = wmma_bf16(a1, bf, acc[1][ni]);
        }
    }

    const int b  = m0 >> 10;
    const int t0 = m0 & 1023;
    #pragma unroll
    for (int ni = 0; ni < 4; ++ni) {
        const int c = c0 + ni * 16 + r;
        const float bv = bias[c];
        #pragma unroll
        for (int mi = 0; mi < 2; ++mi) {
            const long base = ((long)b * CC + c) * NN + t0 + mi * 16 + hi * 8;
            const float4 x0 = *(const float4*)(x + base);
            const float4 x1 = *(const float4*)(x + base + 4);
            float4 w0, w1;
            w0.x = acc[mi][ni][0] + bv + x0.x;  w0.y = acc[mi][ni][1] + bv + x0.y;
            w0.z = acc[mi][ni][2] + bv + x0.z;  w0.w = acc[mi][ni][3] + bv + x0.w;
            w1.x = acc[mi][ni][4] + bv + x1.x;  w1.y = acc[mi][ni][5] + bv + x1.y;
            w1.z = acc[mi][ni][6] + bv + x1.z;  w1.w = acc[mi][ni][7] + bv + x1.w;
            *(float4*)(out + base)     = w0;
            *(float4*)(out + base + 4) = w1;
        }
    }
}

// ---------- host ----------
extern "C" void kernel_launch(void* const* d_in, const int* in_sizes, int n_in,
                              void* d_out, int out_size, void* d_ws, size_t ws_size,
                              hipStream_t stream) {
    const float* x    = (const float*)d_in[0];   // [16,256,32,32]
    const float* Wqkv = (const float*)d_in[1];   // [256,1536]
    const float* bqkv = (const float*)d_in[2];   // [1536]
    const float* Wout = (const float*)d_in[3];   // [512,256]
    const float* bout = (const float*)d_in[4];   // [256]
    float* out = (float*)d_out;

    char* ws = (char*)d_ws;
    __bf16* xs_bf  = (__bf16*)(ws);                         // 16384 x 256   (8 MB)
    __bf16* WqkvT  = (__bf16*)(ws + 8388608);               // 1536 x 256
    __bf16* WoutT  = (__bf16*)(ws + 9175040);               // 256 x 512
    __bf16* Qbf    = (__bf16*)(ws + 9437184);               // [128][1024][64]
    __bf16* Kbf    = (__bf16*)(ws + 26214400);              // [128][1024][64]
    __bf16* Vt     = (__bf16*)(ws + 42991616);              // [128][64][1024]
    __bf16* resb   = (__bf16*)(ws + 59768832);              // 16384 x 512

    // prep transposes/conversions
    transpose_to_bf16<<<dim3(32, 8, 16), dim3(32, 8), 0, stream>>>(
        x, xs_bf, CC, NN, (long)CC * NN, (long)NN * CC);
    transpose_to_bf16<<<dim3(48, 8, 1), dim3(32, 8), 0, stream>>>(
        Wqkv, WqkvT, CC, NQKV, 0, 0);
    transpose_to_bf16<<<dim3(8, 16, 1), dim3(32, 8), 0, stream>>>(
        Wout, WoutT, INNER, CC, 0, 0);

    // QKV projection: 64 m-blocks (256 rows) x 24 n-blocks (64 cols)
    gemm_qkv<<<dim3(64, 24), 256, 0, stream>>>(xs_bf, WqkvT, bqkv, Qbf, Kbf, Vt);

    // flash attention: 128 (b,h) x 16 row-blocks of 64; 4 waves/block
    attn_kernel<<<dim3(128, 16), 128, 0, stream>>>(Qbf, Kbf, Vt, resb);

    // output projection + bias + residual, transposed store
    gemm_out<<<dim3(64, 4), 256, 0, stream>>>(resb, WoutT, bout, x, out);
}
